// Correlation_Block_75350906241093
// MI455X (gfx1250) — matrix operand order repaired and checked by reading, not compiled, not run on hardware
//
#include <hip/hip_runtime.h>
#include <hip/hip_bf16.h>

#define NB_ 32
#define F_  1024
#define D_  512
#define H_  1024
#define EPS_ 1e-5f

typedef __attribute__((ext_vector_type(16))) __bf16 v16bf;
typedef __attribute__((ext_vector_type(8)))  float  v8f;
typedef __attribute__((__vector_size__(4 * sizeof(int)))) int vi4;

#define AS1 __attribute__((address_space(1)))
#define AS3 __attribute__((address_space(3)))

#if __has_builtin(__builtin_amdgcn_global_load_async_to_lds_b128)
#define USE_ASYNC_LDS 1
#endif

struct Pack32 { uint4 lo, hi; };

// A-fragment (16x32 bf16): lane<16 holds M=lane, K {0..7, 16..23}; lane>=16 -> K {8..15, 24..31}
__device__ __forceinline__ v16bf load_fragA(const __bf16* rowbase, int lane) {
  int half = (lane >> 4) & 1;
  int r = lane & 15;
  const char* p = (const char*)(rowbase + r * 40);
  Pack32 pk;
  pk.lo = *(const uint4*)(p + half * 16);
  pk.hi = *(const uint4*)(p + 32 + half * 16);
  return __builtin_bit_cast(v16bf, pk);
}
// B-fragment (32x16 bf16, from Bt rows): lane<16 holds N=lane, K {0..7, 16..23}; lane>=16 -> K {8..15, 24..31}
__device__ __forceinline__ v16bf load_fragB(const __bf16* rowbase, int lane) {
  int half = (lane >> 4) & 1;
  int r = lane & 15;
  const char* p = (const char*)(rowbase + r * 40);
  Pack32 pk;
  pk.lo = *(const uint4*)(p + half * 16);
  pk.hi = *(const uint4*)(p + 32 + half * 16);
  return __builtin_bit_cast(v16bf, pk);
}

#ifdef USE_ASYNC_LDS
template<int N>
__device__ __forceinline__ void wait_async() {
#if __has_builtin(__builtin_amdgcn_s_wait_asynccnt)
  __builtin_amdgcn_s_wait_asynccnt((unsigned short)N);
#else
  if (N == 0) asm volatile("s_wait_asynccnt 0x0" ::: "memory");
  else        asm volatile("s_wait_asynccnt 0x4" ::: "memory");
#endif
}

// Issue async global->LDS copies for one 128x32 bf16 A tile and one B tile.
// Each thread moves 2 x 16B chunks per tile (4 async instructions / thread).
__device__ __forceinline__ void stage_async(const __bf16* Ag, const __bf16* Bg,
                                            __bf16* sA, __bf16* sB,
                                            int tid, int lda, int ldb,
                                            int row0, int col0, int k0) {
#pragma unroll
  for (int c = 0; c < 2; ++c) {
    int id = tid * 2 + c;
    int rr = id >> 2;
    int cc = id & 3;
    const __bf16* ga = Ag + (long long)(row0 + rr) * lda + k0 + cc * 8;
    const __bf16* gb = Bg + (long long)(col0 + rr) * ldb + k0 + cc * 8;
    __bf16* la = sA + rr * 40 + cc * 8;
    __bf16* lb = sB + rr * 40 + cc * 8;
    __builtin_amdgcn_global_load_async_to_lds_b128((AS1 vi4*)ga, (AS3 vi4*)la, 0, 0);
    __builtin_amdgcn_global_load_async_to_lds_b128((AS1 vi4*)gb, (AS3 vi4*)lb, 0, 0);
  }
}
#endif

// C = A(MxK) @ Bt(NxK)^T, bf16 inputs, f32 accumulate via v_wmma_f32_16x16x32_bf16.
// EPI: 0 = f32 + col bias; 1 = softsign -> bf16; 2 = bf16; 3 = f32 + row bias; 4 = f32
template<int EPI>
__global__ __launch_bounds__(256) void gemm_bf16(
    const __bf16* __restrict__ A, const __bf16* __restrict__ Bt, void* __restrict__ Cv,
    const float* __restrict__ bias,
    int M, int Ncols, int K, int lda, int ldb, int ldc,
    long long strA, long long strB, long long strC)
{
  const int z = blockIdx.z;
  A  += (long long)z * strA;
  Bt += (long long)z * strB;
  float*  Cf = (float*)Cv  + (long long)z * strC;
  __bf16* Cb = (__bf16*)Cv + (long long)z * strC;

  const int tid  = threadIdx.x;
  const int lane = tid & 31;
  const int wave = tid >> 5;
  const int wm = wave >> 2;   // 0..1 -> 64-row strip
  const int wn = wave & 3;    // 0..3 -> 32-col strip
  const int row0 = blockIdx.y * 128;
  const int col0 = blockIdx.x * 128;

  v8f acc[4][2] = {};

#ifdef USE_ASYNC_LDS
  // Double-buffered tiles fed by the async global->LDS engine (ASYNCcnt).
  __shared__ __bf16 tA[2][128 * 40];
  __shared__ __bf16 tB[2][128 * 40];

  stage_async(A, Bt, tA[0], tB[0], tid, lda, ldb, row0, col0, 0);
  const int nsteps = K >> 5;
  for (int ks = 0; ks < nsteps; ++ks) {
    const int cur = ks & 1;
    const int nxt = cur ^ 1;
    const bool pre = (ks + 1) < nsteps;
    if (pre) stage_async(A, Bt, tA[nxt], tB[nxt], tid, lda, ldb, row0, col0, (ks + 1) * 32);
    // async ops complete in order: <=4 outstanding means the current slice landed
    if (pre) wait_async<4>(); else wait_async<0>();
    __syncthreads();

    v16bf aF[4], bF[2];
#pragma unroll
    for (int i = 0; i < 4; ++i) aF[i] = load_fragA(&tA[cur][(wm * 64 + i * 16) * 40], lane);
#pragma unroll
    for (int j = 0; j < 2; ++j) bF[j] = load_fragB(&tB[cur][(wn * 32 + j * 16) * 40], lane);
#pragma unroll
    for (int i = 0; i < 4; ++i)
#pragma unroll
      for (int j = 0; j < 2; ++j)
        acc[i][j] = __builtin_amdgcn_wmma_f32_16x16x32_bf16(
            false, aF[i], false, bF[j], (short)0, acc[i][j], false, false);
    __syncthreads();   // all waves done reading buf[cur] before it is refilled
  }
#else
  __shared__ __bf16 tA1[128 * 40];
  __shared__ __bf16 tB1[128 * 40];
  for (int k0 = 0; k0 < K; k0 += 32) {
#pragma unroll
    for (int c = 0; c < 2; ++c) {
      int id = tid * 2 + c;
      int rr = id >> 2;
      int cc = id & 3;
      uint4 da = *(const uint4*)(A  + (long long)(row0 + rr) * lda + k0 + cc * 8);
      uint4 db = *(const uint4*)(Bt + (long long)(col0 + rr) * ldb + k0 + cc * 8);
      *(uint4*)(&tA1[rr * 40 + cc * 8]) = da;
      *(uint4*)(&tB1[rr * 40 + cc * 8]) = db;
    }
    __builtin_prefetch(A  + (long long)(row0 + ((tid * 2) >> 2)) * lda + k0 + 32, 0, 1);
    __builtin_prefetch(Bt + (long long)(col0 + ((tid * 2) >> 2)) * ldb + k0 + 32, 0, 1);
    __syncthreads();

    v16bf aF[4], bF[2];
#pragma unroll
    for (int i = 0; i < 4; ++i) aF[i] = load_fragA(&tA1[(wm * 64 + i * 16) * 40], lane);
#pragma unroll
    for (int j = 0; j < 2; ++j) bF[j] = load_fragB(&tB1[(wn * 32 + j * 16) * 40], lane);
#pragma unroll
    for (int i = 0; i < 4; ++i)
#pragma unroll
      for (int j = 0; j < 2; ++j)
        acc[i][j] = __builtin_amdgcn_wmma_f32_16x16x32_bf16(
            false, aF[i], false, bF[j], (short)0, acc[i][j], false, false);
    __syncthreads();
  }
#endif

  const int half = (lane >> 4) & 1;
  const int cn = lane & 15;
#pragma unroll
  for (int i = 0; i < 4; ++i) {
#pragma unroll
    for (int j = 0; j < 2; ++j) {
      int rbase = row0 + wm * 64 + i * 16 + half * 8;
      int col   = col0 + wn * 32 + j * 16 + cn;
#pragma unroll
      for (int e = 0; e < 8; ++e) {
        int rowg = rbase + e;
        float v = acc[i][j][e];
        long long off = (long long)rowg * ldc + col;
        if (EPI == 0)      Cf[off] = v + bias[col];
        else if (EPI == 1) { v = v / (1.0f + fabsf(v)); Cb[off] = (__bf16)v; }
        else if (EPI == 2) Cb[off] = (__bf16)v;
        else if (EPI == 3) Cf[off] = v + bias[rowg];
        else               Cf[off] = v;
      }
    }
  }
}

__global__ void cvt_bf16(const float* __restrict__ src, __bf16* __restrict__ dst, long long n) {
  long long i = (long long)blockIdx.x * blockDim.x + threadIdx.x;
  if (i < n) dst[i] = (__bf16)src[i];
}

// Per-channel-F stats over (N, L); folds gamma/beta into scale/shift.
__global__ __launch_bounds__(256) void bn_stats(const float* __restrict__ src,
    const float* __restrict__ gamma, const float* __restrict__ beta,
    float* __restrict__ scale, float* __restrict__ shift, int F, int L, int NBatch)
{
  __shared__ float s1[256], s2[256];
  int f = blockIdx.x;
  float a = 0.f, b = 0.f;
  for (int n = 0; n < NBatch; ++n) {
    const float* p = src + ((long long)n * F + f) * L;
    for (int i = threadIdx.x; i < L; i += blockDim.x) {
      float v = p[i];
      a += v; b += v * v;
    }
  }
  s1[threadIdx.x] = a; s2[threadIdx.x] = b;
  __syncthreads();
  for (int s = 128; s > 0; s >>= 1) {
    if (threadIdx.x < s) { s1[threadIdx.x] += s1[threadIdx.x + s]; s2[threadIdx.x] += s2[threadIdx.x + s]; }
    __syncthreads();
  }
  if (threadIdx.x == 0) {
    float cnt = (float)NBatch * (float)L;
    float mean = s1[0] / cnt;
    float var  = s2[0] / cnt - mean * mean;
    float rs   = rsqrtf(var + EPS_);
    float sc   = gamma[f] * rs;
    scale[f] = sc;
    shift[f] = beta[f] - mean * sc;
  }
}

// dst = src*scale[f] + shift[f] (+ resid); writes f32 and/or bf16.
__global__ void bn_apply(const float* __restrict__ src, const float* __restrict__ resid,
    const float* __restrict__ scale, const float* __restrict__ shift,
    float* __restrict__ dstF, __bf16* __restrict__ dstB, int F, int L, long long total)
{
  long long i = (long long)blockIdx.x * blockDim.x + threadIdx.x;
  if (i >= total) return;
  int f = (int)((i / L) % F);
  float v = src[i] * scale[f] + shift[f];
  if (resid) v += resid[i];
  if (dstF) dstF[i] = v;
  if (dstB) dstB[i] = (__bf16)v;
}

__global__ void add2(const float* __restrict__ a, const float* __restrict__ b,
                     float* __restrict__ dst, long long n) {
  long long i = (long long)blockIdx.x * blockDim.x + threadIdx.x;
  if (i < n) dst[i] = a[i] + b[i];
}

// dst[b][c][r] = bf16(src[b][r][c]); grid (C/32, R/32, B), 256 threads (32x8)
template<typename TIN>
__global__ __launch_bounds__(256) void transpose_to_bf16(const TIN* __restrict__ src,
    __bf16* __restrict__ dst, int R, int C)
{
  __shared__ __bf16 tile[32][33];
  long long base = (long long)blockIdx.z * R * C;
  int c0 = blockIdx.x * 32, r0 = blockIdx.y * 32;
  int tx = threadIdx.x & 31, ty = threadIdx.x >> 5;
  for (int i = ty; i < 32; i += 8)
    tile[i][tx] = (__bf16)src[base + (long long)(r0 + i) * C + (c0 + tx)];
  __syncthreads();
  for (int i = ty; i < 32; i += 8)
    dst[base + (long long)(c0 + i) * R + (r0 + tx)] = tile[tx][i];
}

extern "C" void kernel_launch(void* const* d_in, const int* in_sizes, int n_in,
                              void* d_out, int out_size, void* d_ws, size_t ws_size,
                              hipStream_t stream) {
  const float* x   = (const float*)d_in[0];
  const float* W0  = (const float*)d_in[1];
  const float* b0  = (const float*)d_in[2];
  const float* g0  = (const float*)d_in[3];
  const float* be0 = (const float*)d_in[4];
  const float* W1  = (const float*)d_in[5];
  const float* b1  = (const float*)d_in[6];
  const float* g1  = (const float*)d_in[7];
  const float* be1 = (const float*)d_in[8];
  const float* gf  = (const float*)d_in[9];
  const float* bff = (const float*)d_in[10];
  const float* Wc  = (const float*)d_in[11];
  const float* bc  = (const float*)d_in[12];
  const float* go  = (const float*)d_in[13];
  const float* bo  = (const float*)d_in[14];
  float* out = (float*)d_out;

  char* ws = (char*)d_ws;
  const size_t MB = 1ull << 20;
  // Region map (with reuse):
  __bf16* R0b = (__bf16*)(ws);             // 64MB: xb -> wb -> xrTb
  float*  R1f = (float*)(ws + 64 * MB);    // 128MB: v0 -> {v3, t}; t2 reuses v3 slot
  char*   R2  = ws + 192 * MB;             // 64MB: vb -> v2b -> y
  char*   R3  = ws + 256 * MB;             // 64MB: vTb -> xr
  __bf16* W0b = (__bf16*)(ws + 320 * MB);
  __bf16* W1b = (__bf16*)(ws + 322 * MB);
  __bf16* Wcb = (__bf16*)(ws + 324 * MB);
  float* scale0 = (float*)(ws + 326 * MB);
  float* shift0 = scale0 + F_;
  float* scale1 = shift0 + F_;
  float* shift1 = scale1 + F_;
  float* scaleF = shift1 + F_;
  float* shiftF = scaleF + F_;
  float* scaleO = shiftF + F_;
  float* shiftO = scaleO + F_;

  __bf16* xb   = R0b;
  float*  v0   = R1f;
  __bf16* vb   = (__bf16*)R2;
  __bf16* vTb  = (__bf16*)R3;
  __bf16* wb   = R0b;                       // after GEMM1
  __bf16* v2b  = (__bf16*)R2;               // after GEMM2
  float*  v3   = R1f;                       // after bn_apply(v0)
  float*  t    = (float*)(ws + 128 * MB);
  float*  xr   = (float*)R3;                // after GEMM3
  __bf16* xrTb = R0b;                       // after GEMM3
  float*  y    = (float*)R2;                // after GEMM4
  float*  t2   = R1f;                       // after t consumed

  dim3 blk(256);
  auto cdiv = [](long long a, long long b) { return (unsigned)((a + b - 1) / b); };

  const long long nx = (long long)NB_ * F_ * D_;  // 16M
  const long long nv = (long long)NB_ * F_ * H_;  // 32M

  // fp32 -> bf16 operand conversion
  cvt_bf16<<<cdiv(nx, 256), blk, 0, stream>>>(x, xb, nx);
  cvt_bf16<<<cdiv((long long)H_ * D_, 256), blk, 0, stream>>>(W0, W0b, (long long)H_ * D_);
  cvt_bf16<<<cdiv((long long)D_ * H_, 256), blk, 0, stream>>>(W1, W1b, (long long)D_ * H_);
  cvt_bf16<<<cdiv((long long)F_ * F_, 256), blk, 0, stream>>>(Wc, Wcb, (long long)F_ * F_);

  // GEMM1: v0 (NF x H) = x @ W0^T + b0
  gemm_bf16<0><<<dim3(H_ / 128, (NB_ * F_) / 128, 1), blk, 0, stream>>>(
      xb, W0b, v0, b0, NB_ * F_, H_, D_, D_, D_, H_, 0, 0, 0);

  bn_stats<<<F_, blk, 0, stream>>>(v0, g0, be0, scale0, shift0, F_, H_, NB_);
  bn_apply<<<cdiv(nv, 256), blk, 0, stream>>>(v0, nullptr, scale0, shift0, nullptr, vb, F_, H_, nv);

  transpose_to_bf16<__bf16><<<dim3(H_ / 32, F_ / 32, NB_), blk, 0, stream>>>(vb, vTb, F_, H_);

  // GEMM2 (batched): w = softsign(v @ v^T) -> bf16
  gemm_bf16<1><<<dim3(F_ / 128, F_ / 128, NB_), blk, 0, stream>>>(
      vb, vb, wb, nullptr, F_, F_, H_, H_, H_, F_,
      (long long)F_ * H_, (long long)F_ * H_, (long long)F_ * F_);

  // GEMM3 (batched): v2 = w @ v  (Bt = v^T) -> bf16
  gemm_bf16<2><<<dim3(H_ / 128, F_ / 128, NB_), blk, 0, stream>>>(
      wb, vTb, v2b, nullptr, F_, H_, F_, F_, F_, H_,
      (long long)F_ * F_, (long long)H_ * F_, (long long)F_ * H_);

  // GEMM4: v3 (NF x D) = v2 @ W1^T + b1
  gemm_bf16<0><<<dim3(D_ / 128, (NB_ * F_) / 128, 1), blk, 0, stream>>>(
      v2b, W1b, v3, b1, NB_ * F_, D_, H_, H_, H_, D_, 0, 0, 0);

  // BN(v3), t = BN(v3) + x, then feed_norm stats and xr
  bn_stats<<<F_, blk, 0, stream>>>(v3, g1, be1, scale1, shift1, F_, D_, NB_);
  bn_apply<<<cdiv(nx, 256), blk, 0, stream>>>(v3, x, scale1, shift1, t, nullptr, F_, D_, nx);
  bn_stats<<<F_, blk, 0, stream>>>(t, gf, bff, scaleF, shiftF, F_, D_, NB_);
  bn_apply<<<cdiv(nx, 256), blk, 0, stream>>>(t, nullptr, scaleF, shiftF, xr, nullptr, F_, D_, nx);

  transpose_to_bf16<float><<<dim3(D_ / 32, F_ / 32, NB_), blk, 0, stream>>>(xr, xrTb, F_, D_);

  // GEMM5 (batched): y = Wc @ xr + bc (row bias); A shared across batch (strideA = 0)
  gemm_bf16<3><<<dim3(D_ / 128, F_ / 128, NB_), blk, 0, stream>>>(
      Wcb, xrTb, y, bc, F_, D_, F_, F_, F_, D_,
      0ll, (long long)D_ * F_, (long long)F_ * D_);

  // out = BN(y + xr)
  add2<<<cdiv(nx, 256), blk, 0, stream>>>(y, xr, t2, nx);
  bn_stats<<<F_, blk, 0, stream>>>(t2, go, bo, scaleO, shiftO, F_, D_, NB_);
  bn_apply<<<cdiv(nx, 256), blk, 0, stream>>>(t2, nullptr, scaleO, shiftO, out, nullptr, F_, D_, nx);
}
